// SparseMoELanguageModel_24584392802743
// MI455X (gfx1250) — matrix operand order repaired and checked by reading, not compiled, not run on hardware
//
#include <hip/hip_runtime.h>
#include <hip/hip_bf16.h>
#include <math.h>

// ---------------------------------------------------------------------------
// SparseMoE LM forward for gfx1250 (MI455X).  All GEMMs use
// v_wmma_f32_16x16x32_bf16 (bf16 in, f32 accumulate).  Weights stay f32 in
// HBM and are converted to bf16 while staging into LDS (compute-bound, so the
// extra f32 read bytes are free).  The bf16 A-operand tile is staged by the
// Tensor Data Mover (tensor_load_to_lds + s_wait_tensorcnt) when the builtin
// is available.  Wave32 everywhere.
// ---------------------------------------------------------------------------

typedef __attribute__((ext_vector_type(16))) __bf16 bf16x16;
typedef __attribute__((ext_vector_type(8)))  float  f32x8;

#if defined(__gfx1250__) && __has_builtin(__builtin_amdgcn_tensor_load_to_lds) && \
    __has_builtin(__builtin_amdgcn_s_wait_tensorcnt)
#define USE_TDM 1
typedef __attribute__((ext_vector_type(4))) unsigned int u32x4;
typedef __attribute__((ext_vector_type(8))) int          i32x8;
typedef __attribute__((ext_vector_type(4))) int          i32x4;
#else
#define USE_TDM 0
#endif

#define N_LAYER  2
#define N_HEAD   16
#define HEAD_D   64
#define BATCH    2
#define SEQ      2048
#define CDIM     1024
#define NTOK     (BATCH * SEQ)       // 4096
#define NEXP     8
#define CAP      1024
#define FFDIM    4096

static __device__ __forceinline__ float  bf2f(__bf16 x) { return (float)x; }
static __device__ __forceinline__ __bf16 f2bf(float x)  { return (__bf16)x; }

// Load a 32-byte bf16 fragment (16 elems) from two 16B-aligned chunks.
static __device__ __forceinline__ bf16x16 ldFrag2(const __bf16* p0, const __bf16* p1) {
    union { bf16x16 v; uint4 q[2]; } u;
    u.q[0] = *(const uint4*)p0;
    u.q[1] = *(const uint4*)p1;
    return u.v;
}

// ---------------------------------------------------------------------------
// Generic tiled WMMA GEMM:  C[M,N] = act( A[M,K](bf16) @ B[K,N](f32) + bias )
// Block tile 128x64, 8 waves of 32, each wave 32x32 (4 wmma accumulators).
// flags: 1 = ReLU, 2 = bf16 output (else f32), 4 = add f32 residual.
// blockIdx.z batches (MoE experts) via element strides.
// A tile (128x32 bf16) is DMA'd by the TDM with pad_enable producing the
// 80B-padded LDS pitch (D# per ISA 8.3-8.5); B tile is converted f32->bf16
// through VGPRs into a transposed LDS tile.
// ---------------------------------------------------------------------------
__global__ __launch_bounds__(256) void gemm_wmma_bf16(
    const __bf16* __restrict__ A, const float* __restrict__ Bw,
    const float* __restrict__ bias, const float* __restrict__ resid,
    void* __restrict__ Cout,
    int M, int N, int K,
    long long aBatch, long long bBatch, long long biasBatch, long long cBatch,
    int flags)
{
    __shared__ __attribute__((aligned(16))) unsigned char smem[15360];
    unsigned char* smemA = smem;            // 128 rows * 40 bf16 (80B padded)
    unsigned char* smemB = smem + 10240;    //  64 cols * 40 bf16 (B^T layout)

    const int tid  = threadIdx.x;
    const int lane = tid & 31;
    const int wave = tid >> 5;
    const int wm   = wave & 3;          // 4 wave-rows  * 32 = 128
    const int wn   = wave >> 2;         // 2 wave-cols  * 32 = 64
    const int half = lane >> 4;
    const int lr   = lane & 15;

    const int mBase = blockIdx.y * 128;
    const int nBase = blockIdx.x * 64;
    const int z     = blockIdx.z;

    const __bf16* Az = A + (size_t)z * aBatch;
    const float*  Bz = Bw + (size_t)z * bBatch;

    f32x8 zero = {0.f,0.f,0.f,0.f,0.f,0.f,0.f,0.f};
    f32x8 acc[2][2];
    acc[0][0] = zero; acc[0][1] = zero; acc[1][0] = zero; acc[1][1] = zero;

    const int aRow  = tid >> 1;          // 0..127
    const int aHalf = tid & 1;           // 16-elem half of the 32-wide k slab
    const int bK    = tid >> 3;          // 0..31
    const int bN8   = (tid & 7) * 8;     // 0..56
    (void)aRow; (void)aHalf;

    for (int k0 = 0; k0 < K; k0 += 32) {
        // ---- stage A tile: 128 x 32 bf16 ----
#if USE_TDM
        if (wave == 0) {
            // Tensor DMA descriptor (D#): 2D tile, data_size=2B, LDS padding
            // of 2 DWORDs every 16 DWORDs (64B row -> 80B pitch).
            const unsigned long long ga =
                (unsigned long long)(uintptr_t)(Az + (size_t)mBase * K + k0);
            const unsigned ldsA = (unsigned)(uintptr_t)smemA;
            u32x4 g0 = {0u, 0u, 0u, 0u};
            i32x8 g1 = {0, 0, 0, 0, 0, 0, 0, 0};
            i32x4 g2 = {0, 0, 0, 0};
            i32x4 g3 = {0, 0, 0, 0};
            i32x8 g4 = {0, 0, 0, 0, 0, 0, 0, 0};
            g0[0] = 1u;                                      // count=1 (valid user D#)
            g0[1] = ldsA;                                    // lds_addr
            g0[2] = (unsigned)ga;                            // global_addr[31:0]
            g0[3] = (unsigned)((ga >> 32) & 0x01FFFFFFu)     // global_addr[56:32]
                    | (2u << 30);                            // type = 2 ("image")
            g1[0] = (int)((1u << 16)                         // data_size = 2 bytes
                        | (1u << 20)                         // pad_enable
                        | (3u << 22)                         // pad_interval = 16 DW
                        | (1u << 25));                       // pad_amount   = 2 DW
            g1[1] = (int)(((unsigned)K & 0xFFFFu) << 16);    // tensor_dim0[15:0]
            g1[2] = (int)(((unsigned)K >> 16)                // tensor_dim0[31:16]
                        | (((unsigned)M & 0xFFFFu) << 16));  // tensor_dim1[15:0]
            g1[3] = (int)((((unsigned)M >> 16) & 0xFFFFu)    // tensor_dim1[31:16]
                        | (32u << 16));                      // tile_dim0 = 32
            g1[4] = 128;                                     // tile_dim1 = 128 rows
            g1[5] = (int)(unsigned)K;                        // tensor_dim0_stride lo
            __builtin_amdgcn_tensor_load_to_lds(g0, g1, g2, g3, g4, 0);
        }
#else
        {
            const int aR = tid >> 1;
            const int aH = tid & 1;
            const uint4* g = (const uint4*)(Az + (size_t)(mBase + aR) * K + k0 + 16 * aH);
            uint4* s = (uint4*)(smemA + aR * 80 + 32 * aH);
            s[0] = g[0];
            s[1] = g[1];
        }
#endif
        // ---- stage B tile transposed + f32->bf16: Bt[n][k], 64 x 32 ----
        {
            const float4* g = (const float4*)(Bz + (size_t)(k0 + bK) * N + nBase + bN8);
            float4 f0 = g[0], f1 = g[1];
            __bf16* bt = (__bf16*)smemB;
            float v[8] = {f0.x, f0.y, f0.z, f0.w, f1.x, f1.y, f1.z, f1.w};
            #pragma unroll
            for (int j = 0; j < 8; ++j)
                bt[(bN8 + j) * 40 + bK] = f2bf(v[j]);
            if (k0 + 32 < K)
                __builtin_prefetch(Bz + (size_t)(k0 + 32 + bK) * N + nBase + bN8, 0, 1);
        }
#if USE_TDM
        if (wave == 0) __builtin_amdgcn_s_wait_tensorcnt(0);
#endif
        __syncthreads();

        const __bf16* aS = (const __bf16*)smemA;
        const __bf16* bS = (const __bf16*)smemB;
        const int ar0 = wm * 32 + lr;
        const int ar1 = ar0 + 16;
        const int bc0 = wn * 32 + lr;
        const int bc1 = bc0 + 16;
        // A frag: lane half selects k {0..7,16..23} vs {8..15,24..31}
        bf16x16 a0 = ldFrag2(aS + ar0 * 40 + 8 * half, aS + ar0 * 40 + 8 * half + 16);
        bf16x16 a1 = ldFrag2(aS + ar1 * 40 + 8 * half, aS + ar1 * 40 + 8 * half + 16);
        // B frag: lane half selects k {0..7,16..23} vs {8..15,24..31}
        bf16x16 b0 = ldFrag2(bS + bc0 * 40 + 8 * half, bS + bc0 * 40 + 8 * half + 16);
        bf16x16 b1 = ldFrag2(bS + bc1 * 40 + 8 * half, bS + bc1 * 40 + 8 * half + 16);

        acc[0][0] = __builtin_amdgcn_wmma_f32_16x16x32_bf16(false, a0, false, b0, (short)0, acc[0][0], false, false);
        acc[0][1] = __builtin_amdgcn_wmma_f32_16x16x32_bf16(false, a0, false, b1, (short)0, acc[0][1], false, false);
        acc[1][0] = __builtin_amdgcn_wmma_f32_16x16x32_bf16(false, a1, false, b0, (short)0, acc[1][0], false, false);
        acc[1][1] = __builtin_amdgcn_wmma_f32_16x16x32_bf16(false, a1, false, b1, (short)0, acc[1][1], false, false);
        __syncthreads();
    }

    // ---- epilogue ----
    const float* biasZ = bias ? bias + (size_t)z * biasBatch : nullptr;
    #pragma unroll
    for (int fm = 0; fm < 2; ++fm) {
        #pragma unroll
        for (int fn = 0; fn < 2; ++fn) {
            const int col = nBase + wn * 32 + fn * 16 + lr;
            #pragma unroll
            for (int v = 0; v < 8; ++v) {
                const int row = mBase + wm * 32 + fm * 16 + v + 8 * half;
                float val = acc[fm][fn][v];
                if (biasZ)       val += biasZ[col];
                if (flags & 1)   val = fmaxf(val, 0.0f);
                const size_t idx = (size_t)z * cBatch + (size_t)row * N + col;
                if (flags & 4)   val += resid[idx];
                if (flags & 2)   ((__bf16*)Cout)[idx] = f2bf(val);
                else             ((float*)Cout)[idx]  = val;
            }
        }
    }
}

// ---------------------------------------------------------------------------
// Flash attention (full, non-causal).  qkv: [NTOK, 3*CDIM] bf16 as produced by
// the QKV GEMM.  Each block: one (b,h), 128 Q rows (16 per wave); KV tiles of
// 32 with online softmax.  out: [NTOK, CDIM] bf16.
// ---------------------------------------------------------------------------
__global__ __launch_bounds__(256) void attn_fa_wmma(
    const __bf16* __restrict__ qkv, __bf16* __restrict__ out)
{
    __shared__ __attribute__((aligned(16))) unsigned char smemAll[15360];
    __bf16* Vt = (__bf16*)smemAll;                    // [64 d][40] (32 kv + pad)
    unsigned char* pBase = smemAll + 5120;            // 8 waves * 16x40 bf16

    const int tid  = threadIdx.x;
    const int lane = tid & 31;
    const int wave = tid >> 5;
    const int half = lane >> 4;
    const int lr   = lane & 15;

    const int b = blockIdx.z;
    const int h = blockIdx.y;
    const int qRow0 = blockIdx.x * 128 + wave * 16;
    const int hOff  = h * HEAD_D;

    // Q fragments (k slabs d=0..31 and d=32..63), held in registers
    bf16x16 qf[2];
    #pragma unroll
    for (int s = 0; s < 2; ++s) {
        const size_t base = (size_t)(b * SEQ + qRow0 + lr) * (3 * CDIM) + hOff + 32 * s;
        qf[s] = ldFrag2(qkv + base + 8 * half, qkv + base + 16 + 8 * half);
    }

    f32x8 zero = {0.f,0.f,0.f,0.f,0.f,0.f,0.f,0.f};
    f32x8 accO[4];
    #pragma unroll
    for (int i = 0; i < 4; ++i) accO[i] = zero;
    float mrow[8], lrow[8];
    #pragma unroll
    for (int v = 0; v < 8; ++v) { mrow[v] = -1e30f; lrow[v] = 0.0f; }

    const int stageKV = tid >> 3;          // 0..31
    const int stageD8 = (tid & 7) * 8;     // 0..56

    for (int kv0 = 0; kv0 < SEQ; kv0 += 32) {
        // ---- stage V^T tile ----
        {
            const uint4* g = (const uint4*)(qkv + (size_t)(b * SEQ + kv0 + stageKV) * (3 * CDIM)
                                            + 2 * CDIM + hOff + stageD8);
            union { uint4 q; __bf16 e[8]; } u;
            u.q = g[0];
            #pragma unroll
            for (int j = 0; j < 8; ++j)
                Vt[(stageD8 + j) * 40 + stageKV] = u.e[j];
        }
        __syncthreads();

        // ---- S = Q @ K^T  (16 x 32, two N-frags) ----
        f32x8 sfr[2];
        sfr[0] = zero; sfr[1] = zero;
        #pragma unroll
        for (int f = 0; f < 2; ++f) {
            #pragma unroll
            for (int ks = 0; ks < 2; ++ks) {
                const size_t kb = (size_t)(b * SEQ + kv0 + f * 16 + lr) * (3 * CDIM)
                                  + CDIM + hOff + 32 * ks;
                bf16x16 bf = ldFrag2(qkv + kb + 8 * half, qkv + kb + 8 * half + 16);
                sfr[f] = __builtin_amdgcn_wmma_f32_16x16x32_bf16(false, qf[ks], false, bf,
                                                                 (short)0, sfr[f], false, false);
            }
        }
        // scale 1/sqrt(64)
        #pragma unroll
        for (int f = 0; f < 2; ++f)
            #pragma unroll
            for (int v = 0; v < 8; ++v) sfr[f][v] *= 0.125f;

        // ---- online softmax (row stats uniform per 16-lane half) ----
        float scl[8];
        #pragma unroll
        for (int v = 0; v < 8; ++v) {
            float rm = fmaxf(sfr[0][v], sfr[1][v]);
            rm = fmaxf(rm, __shfl_xor(rm, 1));
            rm = fmaxf(rm, __shfl_xor(rm, 2));
            rm = fmaxf(rm, __shfl_xor(rm, 4));
            rm = fmaxf(rm, __shfl_xor(rm, 8));
            float mn = fmaxf(mrow[v], rm);
            scl[v] = __expf(mrow[v] - mn);
            mrow[v] = mn;
        }
        #pragma unroll
        for (int f = 0; f < 2; ++f)
            #pragma unroll
            for (int v = 0; v < 8; ++v) sfr[f][v] = __expf(sfr[f][v] - mrow[v]);
        #pragma unroll
        for (int v = 0; v < 8; ++v) {
            float rs = sfr[0][v] + sfr[1][v];
            rs += __shfl_xor(rs, 1);
            rs += __shfl_xor(rs, 2);
            rs += __shfl_xor(rs, 4);
            rs += __shfl_xor(rs, 8);
            lrow[v] = lrow[v] * scl[v] + rs;
        }
        #pragma unroll
        for (int nf = 0; nf < 4; ++nf)
            #pragma unroll
            for (int v = 0; v < 8; ++v) accO[nf][v] *= scl[v];

        // ---- P (C-layout) -> LDS -> A-layout fragment ----
        __bf16* pT = (__bf16*)(pBase + wave * 1280);
        #pragma unroll
        for (int f = 0; f < 2; ++f)
            #pragma unroll
            for (int v = 0; v < 8; ++v)
                pT[(v + 8 * half) * 40 + f * 16 + lr] = f2bf(sfr[f][v]);
        bf16x16 pf = ldFrag2(pT + lr * 40 + 8 * half, pT + lr * 40 + 8 * half + 16);

        // ---- O += P @ V ----
        #pragma unroll
        for (int nf = 0; nf < 4; ++nf) {
            const __bf16* vp = Vt + (nf * 16 + lr) * 40 + 8 * half;
            bf16x16 vf = ldFrag2(vp, vp + 16);
            accO[nf] = __builtin_amdgcn_wmma_f32_16x16x32_bf16(false, pf, false, vf,
                                                               (short)0, accO[nf], false, false);
        }
        __syncthreads();
    }

    // ---- normalize + store ----
    #pragma unroll
    for (int nf = 0; nf < 4; ++nf) {
        const int col = hOff + nf * 16 + lr;
        #pragma unroll
        for (int v = 0; v < 8; ++v) {
            const int row = qRow0 + v + 8 * half;
            out[(size_t)(b * SEQ + row) * CDIM + col] = f2bf(accO[nf][v] / lrow[v]);
        }
    }
}

// ---------------------------------------------------------------------------
// Embedding:  x[n] = tok_emb[ids[n]] + pos_emb[n % T]
// ---------------------------------------------------------------------------
__global__ __launch_bounds__(256) void embed_kernel(
    const int* __restrict__ ids, const float* __restrict__ tok,
    const float* __restrict__ pos, float* __restrict__ x)
{
    const int n  = blockIdx.x;
    const int id = ids[n];
    const int t  = n & (SEQ - 1);
    for (int c = threadIdx.x; c < CDIM; c += 256)
        x[(size_t)n * CDIM + c] = tok[(size_t)id * CDIM + c] + pos[(size_t)t * CDIM + c];
}

// ---------------------------------------------------------------------------
// LayerNorm (f32 in -> bf16 out), one token row per block.
// ---------------------------------------------------------------------------
__global__ __launch_bounds__(256) void layernorm_bf16(
    const float* __restrict__ x, const float* __restrict__ g,
    const float* __restrict__ b, __bf16* __restrict__ out)
{
    __shared__ float rs[256], rs2[256], stat[2];
    const int row = blockIdx.x;
    const int tid = threadIdx.x;
    float s = 0.f, s2 = 0.f;
    for (int c = tid; c < CDIM; c += 256) {
        float v = x[(size_t)row * CDIM + c];
        s += v; s2 += v * v;
    }
    rs[tid] = s; rs2[tid] = s2;
    __syncthreads();
    for (int off = 128; off > 0; off >>= 1) {
        if (tid < off) { rs[tid] += rs[tid + off]; rs2[tid] += rs2[tid + off]; }
        __syncthreads();
    }
    if (tid == 0) {
        float mean = rs[0] * (1.0f / CDIM);
        float var  = rs2[0] * (1.0f / CDIM) - mean * mean;
        stat[0] = mean;
        stat[1] = rsqrtf(var + 1e-5f);
    }
    __syncthreads();
    const float mean = stat[0], rstd = stat[1];
    for (int c = tid; c < CDIM; c += 256)
        out[(size_t)row * CDIM + c] =
            f2bf((x[(size_t)row * CDIM + c] - mean) * rstd * g[c] + b[c]);
}

// ---------------------------------------------------------------------------
// Router: per-token logits + noisy top-2 gating.
// NOTE: noise uses a threefry-style counter RNG + Box-Muller — deterministic
// but not bit-identical to jax.random (unverifiable without runtime).
// ---------------------------------------------------------------------------
static __device__ __forceinline__ unsigned rotl32(unsigned x, int n) {
    return (x << n) | (x >> (32 - n));
}
static __device__ __forceinline__ uint2 threefry2x32(uint2 key, uint2 ctr) {
    const int R[8] = {13, 15, 26, 6, 17, 29, 16, 24};
    unsigned ks[3] = {key.x, key.y, key.x ^ key.y ^ 0x1BD11BDAu};
    unsigned x0 = ctr.x + ks[0], x1 = ctr.y + ks[1];
    #pragma unroll
    for (int i = 0; i < 5; ++i) {
        #pragma unroll
        for (int j = 0; j < 4; ++j) {
            x0 += x1;
            x1 = rotl32(x1, R[(i & 1) * 4 + j]);
            x1 ^= x0;
        }
        x0 += ks[(i + 1) % 3];
        x1 += ks[(i + 2) % 3] + (unsigned)(i + 1);
    }
    return make_uint2(x0, x1);
}
static __device__ __forceinline__ float rng_normal(unsigned layer, unsigned n, unsigned e) {
    uint2 r = threefry2x32(make_uint2(42u ^ (layer * 0x9E3779B9u), 0x6A09E667u),
                           make_uint2(n, e));
    float u1 = ((r.x >> 8) + 1u) * (1.0f / 16777216.0f);
    float u2 = (r.y >> 8) * (1.0f / 16777216.0f);
    return sqrtf(-2.0f * __logf(u1)) * __cosf(6.28318530718f * u2);
}
static __device__ __forceinline__ float softplus_f(float x) {
    return (x > 20.0f) ? x : log1pf(__expf(x));
}

__global__ __launch_bounds__(256) void router_topk(
    const __bf16* __restrict__ xln,
    const float* __restrict__ rw, const float* __restrict__ rb,
    const float* __restrict__ nw, const float* __restrict__ nb,
    int layer, int* __restrict__ topE, float* __restrict__ topP)
{
    __shared__ float red[256 * 16];
    __shared__ float fin[16];
    const int n = blockIdx.x;
    const int tid = threadIdx.x;
    float p[16];
    #pragma unroll
    for (int i = 0; i < 16; ++i) p[i] = 0.f;
    for (int c = tid; c < CDIM; c += 256) {
        float xv = bf2f(xln[(size_t)n * CDIM + c]);
        #pragma unroll
        for (int e = 0; e < NEXP; ++e) {
            p[e]     += xv * rw[c * NEXP + e];
            p[8 + e] += xv * nw[c * NEXP + e];
        }
    }
    #pragma unroll
    for (int i = 0; i < 16; ++i) red[tid * 16 + i] = p[i];
    __syncthreads();
    if (tid < 16) {
        float s = 0.f;
        for (int t = 0; t < 256; ++t) s += red[t * 16 + tid];
        fin[tid] = s;
    }
    __syncthreads();
    if (tid == 0) {
        float noisy[NEXP];
        #pragma unroll
        for (int e = 0; e < NEXP; ++e) {
            float lg = fin[e] + rb[e];
            float nl = fin[8 + e] + nb[e];
            noisy[e] = lg + rng_normal((unsigned)layer, (unsigned)n, (unsigned)e) * softplus_f(nl);
        }
        int e0 = 0, e1 = 0;
        float v0 = -1e30f, v1 = -1e30f;
        #pragma unroll
        for (int e = 0; e < NEXP; ++e) {
            if (noisy[e] > v0)      { v1 = v0; e1 = e0; v0 = noisy[e]; e0 = e; }
            else if (noisy[e] > v1) { v1 = noisy[e]; e1 = e; }
        }
        float bb  = __expf(v1 - v0);
        float inv = 1.0f / (1.0f + bb);
        topE[2 * n]     = e0;
        topE[2 * n + 1] = e1;
        topP[2 * n]     = inv;
        topP[2 * n + 1] = bb * inv;
    }
}

// Per-expert capacity selection: stable (token-order) prefix via wave ballot.
__global__ void route_select(
    const int* __restrict__ topE, const float* __restrict__ topP,
    int* __restrict__ order, float* __restrict__ gating)
{
    const int e = blockIdx.x;
    const int lane = threadIdx.x;
    int count = 0;
    for (int base = 0; base < NTOK; base += 32) {
        const int tok = base + lane;
        const int t0 = topE[2 * tok], t1 = topE[2 * tok + 1];
        const bool has = (t0 == e) || (t1 == e);
        const unsigned mask = (unsigned)__ballot(has);
        const int pos = count + __popc(mask & ((1u << lane) - 1u));
        if (has && pos < CAP) {
            order[e * CAP + pos]  = tok;
            gating[e * CAP + pos] = (t0 == e) ? topP[2 * tok] : topP[2 * tok + 1];
        }
        count += __popc(mask);
        if (count >= CAP) break;
    }
    const int start = (count < CAP) ? count : CAP;
    for (int i = start + lane; i < CAP; i += 32) {
        order[e * CAP + i]  = 0;
        gating[e * CAP + i] = 0.0f;
    }
}

__global__ __launch_bounds__(256) void moe_gather(
    const __bf16* __restrict__ xln, const int* __restrict__ order,
    __bf16* __restrict__ g)
{
    const int slot = blockIdx.x;           // e*CAP + c
    const int tok  = order[slot];
    for (int c = threadIdx.x; c < CDIM; c += 256)
        g[(size_t)slot * CDIM + c] = xln[(size_t)tok * CDIM + c];
}

__global__ __launch_bounds__(256) void moe_scatter(
    const float* __restrict__ y, const int* __restrict__ order,
    const float* __restrict__ gating, float* __restrict__ x)
{
    const int slot = blockIdx.x;
    const float gt = gating[slot];
    if (gt == 0.0f) return;
    const int tok = order[slot];
    for (int c = threadIdx.x; c < CDIM; c += 256)
        atomicAdd(&x[(size_t)tok * CDIM + c], y[(size_t)slot * CDIM + c] * gt);
}

// Final head: out[n] = dot(xlnf[n], head_w) + head_b  (one wave per token)
__global__ __launch_bounds__(256) void head_mv(
    const __bf16* __restrict__ xln, const float* __restrict__ hw,
    const float* __restrict__ hb, float* __restrict__ out)
{
    const int lane = threadIdx.x & 31;
    const int n = blockIdx.x * 8 + (threadIdx.x >> 5);
    float s = 0.f;
    for (int c = lane; c < CDIM; c += 32)
        s += bf2f(xln[(size_t)n * CDIM + c]) * hw[c];
    s += __shfl_xor(s, 16);
    s += __shfl_xor(s, 8);
    s += __shfl_xor(s, 4);
    s += __shfl_xor(s, 2);
    s += __shfl_xor(s, 1);
    if (lane == 0) out[n] = s + hb[0];
}

// ---------------------------------------------------------------------------
// Host orchestration
// ---------------------------------------------------------------------------
extern "C" void kernel_launch(void* const* d_in, const int* in_sizes, int n_in,
                              void* d_out, int out_size, void* d_ws, size_t ws_size,
                              hipStream_t stream)
{
    const int*   input_ids = (const int*)  d_in[0];
    const float* tok_emb   = (const float*)d_in[1];
    const float* pos_emb   = (const float*)d_in[2];
    const float* ln1_g     = (const float*)d_in[3];
    const float* ln1_b     = (const float*)d_in[4];
    const float* ln2_g     = (const float*)d_in[5];
    const float* ln2_b     = (const float*)d_in[6];
    const float* qkv_w     = (const float*)d_in[7];
    const float* out_w     = (const float*)d_in[8];
    const float* route_w   = (const float*)d_in[9];
    const float* route_b   = (const float*)d_in[10];
    const float* noise_w   = (const float*)d_in[11];
    const float* noise_b   = (const float*)d_in[12];
    const float* w1        = (const float*)d_in[13];
    const float* b1        = (const float*)d_in[14];
    const float* w2        = (const float*)d_in[15];
    const float* b2        = (const float*)d_in[16];
    const float* lnf_g     = (const float*)d_in[17];
    const float* lnf_b     = (const float*)d_in[18];
    const float* head_w    = (const float*)d_in[19];
    const float* head_b    = (const float*)d_in[20];

    // workspace layout (bytes, 256-aligned); total ~176 MB
    size_t off = 0;
    auto alloc = [&](size_t bytes) { size_t r = off; off += (bytes + 255) & ~(size_t)255; return r; };
    unsigned char* ws = (unsigned char*)d_ws;
    float*  X      = (float*) (ws + alloc((size_t)NTOK * CDIM * 4));
    __bf16* XLN    = (__bf16*)(ws + alloc((size_t)NTOK * CDIM * 2));
    __bf16* QKV    = (__bf16*)(ws + alloc((size_t)NTOK * 3 * CDIM * 2));
    __bf16* ATT    = (__bf16*)(ws + alloc((size_t)NTOK * CDIM * 2));
    int*    TOPE   = (int*)   (ws + alloc((size_t)NTOK * 2 * 4));
    float*  TOPP   = (float*) (ws + alloc((size_t)NTOK * 2 * 4));
    int*    ORDER  = (int*)   (ws + alloc((size_t)NEXP * CAP * 4));
    float*  GATING = (float*) (ws + alloc((size_t)NEXP * CAP * 4));
    __bf16* GBUF   = (__bf16*)(ws + alloc((size_t)NEXP * CAP * CDIM * 2));
    __bf16* HBUF   = (__bf16*)(ws + alloc((size_t)NEXP * CAP * FFDIM * 2));
    float*  YBUF   = (float*) (ws + alloc((size_t)NEXP * CAP * CDIM * 4));
    (void)ws_size; (void)n_in; (void)in_sizes; (void)out_size;

    embed_kernel<<<NTOK, 256, 0, stream>>>(input_ids, tok_emb, pos_emb, X);

    for (int l = 0; l < N_LAYER; ++l) {
        // ---- attention block ----
        layernorm_bf16<<<NTOK, 256, 0, stream>>>(X, ln1_g + l * CDIM, ln1_b + l * CDIM, XLN);

        gemm_wmma_bf16<<<dim3(3 * CDIM / 64, NTOK / 128, 1), 256, 0, stream>>>(
            XLN, qkv_w + (size_t)l * CDIM * 3 * CDIM, nullptr, nullptr, QKV,
            NTOK, 3 * CDIM, CDIM, 0, 0, 0, 0, /*flags=*/2);

        attn_fa_wmma<<<dim3(SEQ / 128, N_HEAD, BATCH), 256, 0, stream>>>(QKV, ATT);

        gemm_wmma_bf16<<<dim3(CDIM / 64, NTOK / 128, 1), 256, 0, stream>>>(
            ATT, out_w + (size_t)l * CDIM * CDIM, nullptr, X, X,
            NTOK, CDIM, CDIM, 0, 0, 0, 0, /*flags=*/4);

        // ---- MoE block ----
        layernorm_bf16<<<NTOK, 256, 0, stream>>>(X, ln2_g + l * CDIM, ln2_b + l * CDIM, XLN);

        router_topk<<<NTOK, 256, 0, stream>>>(
            XLN, route_w + (size_t)l * CDIM * NEXP, route_b + l * NEXP,
            noise_w + (size_t)l * CDIM * NEXP, noise_b + l * NEXP, l, TOPE, TOPP);

        route_select<<<NEXP, 32, 0, stream>>>(TOPE, TOPP, ORDER, GATING);

        moe_gather<<<NEXP * CAP, 256, 0, stream>>>(XLN, ORDER, GBUF);

        gemm_wmma_bf16<<<dim3(FFDIM / 64, CAP / 128, NEXP), 256, 0, stream>>>(
            GBUF, w1 + (size_t)l * NEXP * CDIM * FFDIM, b1 + (size_t)l * NEXP * FFDIM,
            nullptr, HBUF, CAP, FFDIM, CDIM,
            (long long)CAP * CDIM, (long long)CDIM * FFDIM, FFDIM,
            (long long)CAP * FFDIM, /*flags=*/1 | 2);

        gemm_wmma_bf16<<<dim3(CDIM / 64, CAP / 128, NEXP), 256, 0, stream>>>(
            HBUF, w2 + (size_t)l * NEXP * FFDIM * CDIM, b2 + (size_t)l * NEXP * CDIM,
            nullptr, YBUF, CAP, CDIM, FFDIM,
            (long long)CAP * FFDIM, (long long)FFDIM * CDIM, CDIM,
            (long long)CAP * CDIM, /*flags=*/0);

        moe_scatter<<<NEXP * CAP, 256, 0, stream>>>(YBUF, ORDER, GATING, X);
    }

    layernorm_bf16<<<NTOK, 256, 0, stream>>>(X, lnf_g, lnf_b, XLN);
    head_mv<<<NTOK / 8, 256, 0, stream>>>(XLN, head_w, head_b, (float*)d_out);
}
